// NeuralODEModel_49950469653084
// MI455X (gfx1250) — compile-verified, matrix operand order checked by reading, not run on hardware
//
#include <hip/hip_runtime.h>
#include <cmath>

// ---------------------------------------------------------------------------
// Neural ODE on MI455X (gfx1250): 200 Euler steps of a 64->256->256->64 MLP.
// Weights staged once into LDS as pre-swizzled f16 WMMA A-fragments (W^T),
// f32 state resident in registers, GEMMs computed transposed (y^T = W^T x^T)
// so each layer's C-tiles ARE the next layer's B-fragments.
// A per-step asm memory clobber pins the weight-fragment loads inside the
// step loop (prevents LICM from hoisting ~450 VGPRs of weights -> spills).
// ---------------------------------------------------------------------------

typedef __attribute__((ext_vector_type(16))) _Float16 v16h;
typedef __attribute__((ext_vector_type(8)))  float    v8f;
typedef __attribute__((ext_vector_type(4)))  float    v4f;

#define BATCH   4096
#define SDIM    64
#define HDIM    256
#define TSTEPS  200
#define DT_F    0.01f

#define WAVES_PER_WG  2
#define BLOCK_THREADS (WAVES_PER_WG * 32)
#define GRID_WGS      (BATCH / (WAVES_PER_WG * 16))   // 128

// fast tanh if the gfx1250 trans-op builtin exists, else libm
#if __has_builtin(__builtin_amdgcn_tanhf)
#define TANHF(x) __builtin_amdgcn_tanhf(x)
#elif __has_builtin(__builtin_amdgcn_tanh_f32)
#define TANHF(x) __builtin_amdgcn_tanh_f32(x)
#else
#define TANHF(x) tanhf(x)
#endif

// ---- LDS layout (f16 elements for weight fragments, then f32 biases) ------
// Fragment = 16x16(N-tile) x 32(K-chunk) of W^T in per-lane A order:
//   elem index = frag*512 + lane*16 + j
//   n (output col of W) = ntile*16 + (lane&15)
//   k (input row of W)  = kchunk*32 + (j>>3)*16 + (j&7) + (lane>>4)*8
#define W1F_OFF   0              // (64/32)*(256/16)=32 frags   -> 16384 f16
#define W2F_OFF   16384          // (256/32)*(256/16)=128 frags -> 65536 f16
#define W3F_OFF   81920          // (256/32)*(64/16)=32 frags   -> 16384 f16
#define WF_TOTAL  98304          // f16 elems (196608 bytes)
#define B1_BYTE   (WF_TOTAL * 2)
#define B2_BYTE   (B1_BYTE + 1024)
#define B3_BYTE   (B2_BYTE + 1024)
#define SMEM_BYTES (B3_BYTE + 256)   // 198912 bytes < 320KB WGP LDS

__device__ __forceinline__ void stage_wfrags(const float* __restrict__ W,
                                             int K, int N, _Float16* dst,
                                             int tid, int nthr) {
  const int nnt   = N >> 4;
  const int total = (K >> 5) * nnt * 512;
  for (int idx = tid; idx < total; idx += nthr) {
    const int j    = idx & 15;
    const int lane = (idx >> 4) & 31;
    const int f    = idx >> 9;
    const int kc   = f / nnt;
    const int nt   = f - kc * nnt;
    const int n    = (nt << 4) + (lane & 15);
    const int k    = (kc << 5) + ((j >> 3) << 4) + (j & 7) + ((lane >> 4) << 3);
    dst[idx] = (_Float16)W[k * N + n];
  }
}

#define WMMA_F16(A, Bv, Cv) \
  __builtin_amdgcn_wmma_f32_16x16x32_f16(false, (A), false, (Bv), (short)0, (Cv), false, false)

// load a v8f bias fragment (8 consecutive f32, 32B aligned) from LDS
__device__ __forceinline__ v8f bias_frag(const float* p) {
  const v4f* q = (const v4f*)p;
  v4f a = q[0], b = q[1];
  v8f c;
  c[0] = a.x; c[1] = a.y; c[2] = a.z; c[3] = a.w;
  c[4] = b.x; c[5] = b.y; c[6] = b.z; c[7] = b.w;
  return c;
}

// streaming store of one 8-float slice (two 16B non-temporal stores)
__device__ __forceinline__ void stream_store8(float* p, const float s[8]) {
  v4f a; a.x = s[0]; a.y = s[1]; a.z = s[2]; a.w = s[3];
  v4f b; b.x = s[4]; b.y = s[5]; b.z = s[6]; b.w = s[7];
  __builtin_nontemporal_store(a, (v4f*)p);
  __builtin_nontemporal_store(b, (v4f*)(p + 4));
}

__global__ __launch_bounds__(BLOCK_THREADS)
void neural_ode_kernel(const float* __restrict__ x0,
                       const float* __restrict__ W1, const float* __restrict__ b1,
                       const float* __restrict__ W2, const float* __restrict__ b2,
                       const float* __restrict__ W3, const float* __restrict__ b3,
                       const float* __restrict__ dt_scale,
                       float* __restrict__ out) {
  extern __shared__ __align__(128) char smem[];
  _Float16* wf  = (_Float16*)smem;
  float*    lb1 = (float*)(smem + B1_BYTE);
  float*    lb2 = (float*)(smem + B2_BYTE);
  float*    lb3 = (float*)(smem + B3_BYTE);

  const int tid = threadIdx.x;

  // ---- one-time weight/bias staging (amortized over 200 steps) ----
  stage_wfrags(W1, SDIM, HDIM, wf + W1F_OFF, tid, BLOCK_THREADS);
  stage_wfrags(W2, HDIM, HDIM, wf + W2F_OFF, tid, BLOCK_THREADS);
  stage_wfrags(W3, HDIM, SDIM, wf + W3F_OFF, tid, BLOCK_THREADS);
  for (int i = tid; i < HDIM; i += BLOCK_THREADS) { lb1[i] = b1[i]; lb2[i] = b2[i]; }
  for (int i = tid; i < SDIM; i += BLOCK_THREADS) { lb3[i] = b3[i]; }
  __syncthreads();

  const int wave = tid >> 5;
  const int lane = tid & 31;
  const int hi8  = (lane >> 4) << 3;      // K/M half offset for this lane
  const int col  = lane & 15;             // batch column within the tile
  const int row  = (blockIdx.x * WAVES_PER_WG + wave) * 16 + col;  // batch row
  const float scale = dt_scale[0] * DT_F;

  // ---- f32 state, resident in registers in C-tile layout:
  //      xs[nt][v] = x[row][nt*16 + hi8 + v]
  float xs[4][8];
#pragma unroll
  for (int nt = 0; nt < 4; ++nt) {
    const v4f* p = (const v4f*)(x0 + (size_t)row * SDIM + nt * 16 + hi8);
    v4f a = p[0], b = p[1];
    xs[nt][0] = a.x; xs[nt][1] = a.y; xs[nt][2] = a.z; xs[nt][3] = a.w;
    xs[nt][4] = b.x; xs[nt][5] = b.y; xs[nt][6] = b.z; xs[nt][7] = b.w;
  }

  // ---- trajectory slot t=0 (streaming / non-temporal) ----
  {
    float* o = out + (size_t)row * (TSTEPS + 1) * SDIM;
#pragma unroll
    for (int nt = 0; nt < 4; ++nt) stream_store8(o + nt * 16 + hi8, xs[nt]);
  }

  // ---- 200 sequential Euler steps ----
  for (int t = 1; t <= TSTEPS; ++t) {
    // Pin all LDS weight loads inside this iteration: without this, LICM
    // hoists ~450 VGPRs of loop-invariant weight fragments and spills.
    __asm__ volatile("" ::: "memory");

    // x^T B-fragments straight from register state (pure VALU cvt/pack)
    v16h xb[2];
#pragma unroll
    for (int nt = 0; nt < 4; ++nt) {
      const int kc = nt >> 1, r = (nt & 1) << 3;
#pragma unroll
      for (int v = 0; v < 8; ++v) xb[kc][r + v] = (_Float16)xs[nt][v];
    }

    // ---- layer 1: h1^T = tanh(W1^T x^T + b1), H=256 -> 16 N-tiles, K=64 ----
    v16h h1[8];
#pragma unroll
    for (int nt = 0; nt < 16; ++nt) {
      v8f c = bias_frag(lb1 + nt * 16 + hi8);
#pragma unroll
      for (int kc = 0; kc < 2; ++kc) {
        v16h a = *(const v16h*)(wf + W1F_OFF + (kc * 16 + nt) * 512 + lane * 16);
        c = WMMA_F16(a, xb[kc], c);
      }
      const int kc2 = nt >> 1, r = (nt & 1) << 3;
#pragma unroll
      for (int v = 0; v < 8; ++v) h1[kc2][r + v] = (_Float16)TANHF(c[v]);
    }

    // ---- layer 2: h2^T = tanh(W2^T h1^T + b2), K=256 -> 8 K-chunks ----
    v16h h2[8];
#pragma unroll
    for (int nt = 0; nt < 16; ++nt) {
      v8f c = bias_frag(lb2 + nt * 16 + hi8);
#pragma unroll
      for (int kc = 0; kc < 8; ++kc) {
        v16h a = *(const v16h*)(wf + W2F_OFF + (kc * 16 + nt) * 512 + lane * 16);
        c = WMMA_F16(a, h1[kc], c);
      }
      const int kc2 = nt >> 1, r = (nt & 1) << 3;
#pragma unroll
      for (int v = 0; v < 8; ++v) h2[kc2][r + v] = (_Float16)TANHF(c[v]);
    }

    // ---- layer 3: dx^T = W3^T h2^T + b3; Euler update + trajectory store ----
    float* o = out + ((size_t)row * (TSTEPS + 1) + t) * SDIM;
#pragma unroll
    for (int nt = 0; nt < 4; ++nt) {
      v8f c = bias_frag(lb3 + nt * 16 + hi8);
#pragma unroll
      for (int kc = 0; kc < 8; ++kc) {
        v16h a = *(const v16h*)(wf + W3F_OFF + (kc * 4 + nt) * 512 + lane * 16);
        c = WMMA_F16(a, h2[kc], c);
      }
#pragma unroll
      for (int v = 0; v < 8; ++v) xs[nt][v] = fmaf(c[v], scale, xs[nt][v]);
      stream_store8(o + nt * 16 + hi8, xs[nt]);
    }
  }
}

extern "C" void kernel_launch(void* const* d_in, const int* in_sizes, int n_in,
                              void* d_out, int out_size, void* d_ws, size_t ws_size,
                              hipStream_t stream) {
  (void)in_sizes; (void)n_in; (void)out_size; (void)d_ws; (void)ws_size;
  const float* x0       = (const float*)d_in[0];
  const float* W1       = (const float*)d_in[1];
  const float* b1       = (const float*)d_in[2];
  const float* W2       = (const float*)d_in[3];
  const float* b2       = (const float*)d_in[4];
  const float* W3       = (const float*)d_in[5];
  const float* b3       = (const float*)d_in[6];
  const float* dt_scale = (const float*)d_in[7];
  float* out = (float*)d_out;

  neural_ode_kernel<<<dim3(GRID_WGS), dim3(BLOCK_THREADS), SMEM_BYTES, stream>>>(
      x0, W1, b1, W2, b2, W3, b3, dt_scale, out);
}
